// DogeCDMoME_49787260895689
// MI455X (gfx1250) — matrix operand order repaired and checked by reading, not compiled, not run on hardware
//
#include <hip/hip_runtime.h>
#include <hip/hip_bf16.h>

// ---------------------------------------------------------------------------
// Types for CDNA5 WMMA
// ---------------------------------------------------------------------------
typedef __attribute__((ext_vector_type(16))) __bf16 v16bf;
typedef __attribute__((ext_vector_type(8)))  float  v8f;

union FragBF {
    v16bf v;
    uint4 u[2];
};

__device__ __forceinline__ unsigned short f32_to_bf16(float f) {
    unsigned int x = __float_as_uint(f);
    unsigned int r = x + 0x7FFFu + ((x >> 16) & 1u);   // round-to-nearest-even
    return (unsigned short)(r >> 16);
}

__device__ __forceinline__ unsigned int pack_bf16x2(float lo, float hi) {
    return (unsigned int)f32_to_bf16(lo) | ((unsigned int)f32_to_bf16(hi) << 16);
}

// ---------------------------------------------------------------------------
// K0: f32 -> bf16 conversion (vectorized)
// ---------------------------------------------------------------------------
__global__ void f32_to_bf16_kernel(const float* __restrict__ in,
                                   unsigned int* __restrict__ out, int n4) {
    int i = blockIdx.x * blockDim.x + threadIdx.x;
    if (i >= n4) return;
    float4 v = ((const float4*)in)[i];
    out[(size_t)i * 2 + 0] = pack_bf16x2(v.x, v.y);
    out[(size_t)i * 2 + 1] = pack_bf16x2(v.z, v.w);
}

// ---------------------------------------------------------------------------
// K_M: fold W_q into keys:
//   Mm[c][p*512 + head*128 + k] = sum_d Wq[c][p*2048+head*512+d] * keys[head][k][p][d]
// Wq: [1024, 4096], keys: [4,128,2,512], Mm: [1024,1024]
// ---------------------------------------------------------------------------
__global__ void build_m_kernel(const float* __restrict__ Wq,
                               const float* __restrict__ keys,
                               float* __restrict__ Mm) {
    int col = blockIdx.x;                              // 0..1023
    int c   = threadIdx.x + blockIdx.y * blockDim.x;   // 0..1023
    int p    = col >> 9;
    int head = (col >> 7) & 3;
    int kk   = col & 127;
    const float* wrow = Wq + (size_t)c * 4096 + p * 2048 + head * 512;
    const float* krow = keys + ((size_t)(head * 128 + kk) * 2 + p) * 512;
    float s = 0.f;
#pragma unroll 8
    for (int d = 0; d < 512; ++d) s += wrow[d] * krow[d];
    Mm[(size_t)c * 1024 + col] = s;
}

// ---------------------------------------------------------------------------
// Tiled bf16 WMMA GEMM (fully static shapes):
//   C[M,N] = A[M,K](bf16) * B[K,N](f32 -> bf16)
// Block tile 128x256, 8 waves (2x4) each computing 64x64 (16 WMMAs/K-step).
// Optional SiLU epilogue, f32 and/or bf16 output.
// ---------------------------------------------------------------------------
#define TILE_M 128
#define TILE_N 256
#define TILE_K 32
#define LDSS   40   // padded bf16 row stride (80B, 16B aligned, bank-skewed)

template <int M, int N, int K, bool SILU, bool WF32, bool WBF16>
__global__ __launch_bounds__(256)
void gemm_bf16_wmma(const unsigned short* __restrict__ A,  // bf16 [M][K]
                    const float* __restrict__ B,           // f32  [K][N]
                    float* __restrict__ Cf,
                    unsigned short* __restrict__ Cb) {
    __shared__ unsigned short As[TILE_M * LDSS];       // 10.0 KB
    __shared__ unsigned short Bs[TILE_N * LDSS];       // 20.0 KB, transposed [n][k]

    const int tid  = threadIdx.x;
    const int lane = tid & 31;
    const int wid  = tid >> 5;
    const int m0 = blockIdx.y * TILE_M;
    const int n0 = blockIdx.x * TILE_N;
    const int waveM = (wid >> 2) * 64;                 // 0 or 64
    const int waveN = (wid & 3) * 64;                  // 0,64,128,192

    v8f acc[4][4];
#pragma unroll
    for (int i = 0; i < 4; ++i)
#pragma unroll
        for (int j = 0; j < 4; ++j)
#pragma unroll
            for (int e = 0; e < 8; ++e) acc[i][j][e] = 0.f;

    const int row16 = lane & 15;
    const int kA    = (lane >> 4) * 8;                 // A frag: quirky K split
    const int kB    = (lane >> 4) * 8;                 // B frag: same K split as A

    // static base pointers for staging (offsets become immediates)
    const unsigned short* aBase = A + (size_t)(m0 + (tid >> 2)) * K + (tid & 3) * 8;
    const float*          bBase = B + n0 + tid;

    for (int k0 = 0; k0 < K; k0 += TILE_K) {
        // ---- stage A tile: 128 x 32 bf16 (512 x uint4), 2 chunks/thread ----
#pragma unroll
        for (int i = 0; i < 2; ++i) {
            // rows tid>>2 and (tid>>2)+64 ; col segment (tid&3)*8
            uint4 v = *(const uint4*)(aBase + (size_t)i * 64 * K + k0);
            int r = (tid >> 2) + i * 64;
            *(uint4*)(As + r * LDSS + (tid & 3) * 8) = v;
        }
        // ---- stage B tile: one N-column per thread; coalesced b32 loads with
        //      immediate offsets, pack bf16 pairs, 4x ds_store_b128 ----
        {
            const float* bcol = bBase + (size_t)k0 * N;
            unsigned int w[16];
#pragma unroll
            for (int kk = 0; kk < 32; kk += 2) {
                float f0 = bcol[(size_t)kk * N];           // offset = kk*N*4 (imm)
                float f1 = bcol[(size_t)(kk + 1) * N];
                w[kk >> 1] = pack_bf16x2(f0, f1);
            }
            uint4* dst = (uint4*)(Bs + tid * LDSS);        // 80B pitch, 16B aligned
#pragma unroll
            for (int q = 0; q < 4; ++q)
                dst[q] = make_uint4(w[q * 4], w[q * 4 + 1], w[q * 4 + 2], w[q * 4 + 3]);
        }
        // prefetch next K tile into cache hierarchy (global_prefetch_b8)
        if (k0 + TILE_K < K) {
            __builtin_prefetch(aBase + (size_t)k0 + TILE_K, 0, 1);
            __builtin_prefetch(bBase + (size_t)(k0 + TILE_K) * N, 0, 1);
        }
        __syncthreads();

        // ---- load fragments from LDS ----
        v16bf afr[4], bfr[4];
#pragma unroll
        for (int mt = 0; mt < 4; ++mt) {
            const unsigned short* p = As + (waveM + mt * 16 + row16) * LDSS + kA;
            FragBF f;
            f.u[0] = *(const uint4*)(p);        // K = kA .. kA+7
            f.u[1] = *(const uint4*)(p + 16);   // K = kA+16 .. kA+23
            afr[mt] = f.v;
        }
#pragma unroll
        for (int nt = 0; nt < 4; ++nt) {
            const unsigned short* p = Bs + (waveN + nt * 16 + row16) * LDSS + kB;
            FragBF f;
            f.u[0] = *(const uint4*)(p);        // K = kB .. kB+7
            f.u[1] = *(const uint4*)(p + 16);   // K = kB+16 .. kB+23
            bfr[nt] = f.v;
        }

        // ---- 16 WMMAs per wave per K-step ----
#pragma unroll
        for (int mt = 0; mt < 4; ++mt)
#pragma unroll
            for (int nt = 0; nt < 4; ++nt)
                acc[mt][nt] = __builtin_amdgcn_wmma_f32_16x16x32_bf16(
                    false, afr[mt], false, bfr[nt],
                    (short)0, acc[mt][nt], false, false);
        __syncthreads();
    }

    // ---- epilogue: C layout VGPR j -> M = j + (lane>=16 ? 8 : 0), N = lane&15 ----
    const int rofs = (lane >> 4) * 8;
#pragma unroll
    for (int mt = 0; mt < 4; ++mt) {
#pragma unroll
        for (int nt = 0; nt < 4; ++nt) {
            int gm_base = m0 + waveM + mt * 16 + rofs;
            int gn = n0 + waveN + nt * 16 + (lane & 15);
#pragma unroll
            for (int j = 0; j < 8; ++j) {
                float v = acc[mt][nt][j];
                if (SILU) v = v / (1.f + __expf(-v));
                size_t off = (size_t)(gm_base + j) * N + gn;
                if (WF32)  Cf[off] = v;
                if (WBF16) Cb[off] = f32_to_bf16(v);
            }
        }
    }
}

// ---------------------------------------------------------------------------
// K4: product-key two-stage top-k + softmax.  One thread per (t, head).
// sim: [2048][1024] with col = p*512 + head*128 + k
// ---------------------------------------------------------------------------
__device__ __forceinline__ void insert_top8(float v, int id, float* vals, int* ids) {
    if (v <= vals[7]) return;                 // strict >: earlier index wins ties
    int pos = 7;
    while (pos > 0 && v > vals[pos - 1]) {
        vals[pos] = vals[pos - 1];
        ids[pos]  = ids[pos - 1];
        --pos;
    }
    vals[pos] = v;
    ids[pos]  = id;
}

__global__ void topk_kernel(const float* __restrict__ sim,
                            int* __restrict__ idxOut,
                            float* __restrict__ wOut) {
    int gid = blockIdx.x * blockDim.x + threadIdx.x;   // 0..8191
    int t = gid >> 2;
    int head = gid & 3;
    const float* sx = sim + (size_t)t * 1024 + head * 128;        // p = 0
    const float* sy = sx + 512;                                   // p = 1

    float vx[8], vy[8];
    int ix[8], iy[8];
#pragma unroll
    for (int i = 0; i < 8; ++i) { vx[i] = -INFINITY; vy[i] = -INFINITY; ix[i] = 0; iy[i] = 0; }
    for (int k = 0; k < 128; ++k) insert_top8(sx[k], k, vx, ix);
    for (int k = 0; k < 128; ++k) insert_top8(sy[k], k, vy, iy);

    float vs[8]; int is[8];
#pragma unroll
    for (int i = 0; i < 8; ++i) { vs[i] = -INFINITY; is[i] = 0; }
    for (int i = 0; i < 8; ++i)
        for (int j = 0; j < 8; ++j)
            insert_top8(vx[i] + vy[j], ix[i] * 128 + iy[j], vs, is);

    // softmax over 8 (vs sorted descending => vs[0] is max)
    float m = vs[0], sum = 0.f, ex[8];
#pragma unroll
    for (int i = 0; i < 8; ++i) { ex[i] = __expf(vs[i] - m); sum += ex[i]; }
    float inv = 1.f / sum;
    int base = t * 32 + head * 8;
#pragma unroll
    for (int i = 0; i < 8; ++i) {
        idxOut[base + i] = is[i];
        wOut[base + i]   = ex[i] * inv;
    }
}

// ---------------------------------------------------------------------------
// K5: expert readout.  One block per token t.
// out[t,:] = sum_{e in 32} silu(h[t]·up[idx_e]) * w_e * down[idx_e]
// ---------------------------------------------------------------------------
__global__ __launch_bounds__(256)
void moe_out_kernel(const float* __restrict__ h,      // [2048][1024]
                    const float* __restrict__ upE,    // [16384][1024]
                    const float* __restrict__ dnE,    // [16384][2048]
                    const int* __restrict__ idx,      // [2048][32]
                    const float* __restrict__ w,      // [2048][32]
                    float* __restrict__ out) {        // [2048][2048]
    __shared__ float hs[1024];
    __shared__ float red[9];
    const int t = blockIdx.x;
    const int tid = threadIdx.x;

    for (int i = tid; i < 1024; i += 256) hs[i] = h[(size_t)t * 1024 + i];
    float acc[8];
#pragma unroll
    for (int j = 0; j < 8; ++j) acc[j] = 0.f;
    __syncthreads();

    for (int e = 0; e < 32; ++e) {
        int id   = idx[t * 32 + e];
        float wt = w[t * 32 + e];
        const float* ue = upE + (size_t)id * 1024;
        float part = 0.f;
#pragma unroll
        for (int i = tid; i < 1024; i += 256) part += hs[i] * ue[i];
        // wave32 reduce
#pragma unroll
        for (int o = 16; o > 0; o >>= 1) part += __shfl_down(part, o, 32);
        if ((tid & 31) == 0) red[tid >> 5] = part;
        __syncthreads();
        if (tid == 0) {
            float s = 0.f;
#pragma unroll
            for (int i = 0; i < 8; ++i) s += red[i];
            red[8] = s;
        }
        __syncthreads();
        float dot = red[8];
        __syncthreads();
        float g = (dot / (1.f + __expf(-dot))) * wt;
        const float* de = dnE + (size_t)id * 2048;
#pragma unroll
        for (int j = 0; j < 8; ++j) acc[j] += g * de[tid + j * 256];
    }
#pragma unroll
    for (int j = 0; j < 8; ++j) out[(size_t)t * 2048 + tid + j * 256] = acc[j];
}

// ---------------------------------------------------------------------------
// Launcher
// ---------------------------------------------------------------------------
extern "C" void kernel_launch(void* const* d_in, const int* in_sizes, int n_in,
                              void* d_out, int out_size, void* d_ws, size_t ws_size,
                              hipStream_t stream) {
    (void)in_sizes; (void)n_in; (void)out_size; (void)ws_size;
    const float* X    = (const float*)d_in[0];   // [2048,2048]
    const float* Wup  = (const float*)d_in[1];   // [2048,8192]
    const float* Wdn  = (const float*)d_in[2];   // [8192,1024]
    const float* Wq   = (const float*)d_in[3];   // [1024,4096]
    const float* keys = (const float*)d_in[4];   // [4,128,2,512]
    const float* upE  = (const float*)d_in[5];   // [16384,1024]
    const float* dnE  = (const float*)d_in[6];   // [16384,2048]
    float* out = (float*)d_out;

    char* ws = (char*)d_ws;
    const size_t MB = 1024u * 1024u;
    unsigned short* Xb  = (unsigned short*)(ws + 0);          //  8 MB bf16 X
    unsigned short* U   = (unsigned short*)(ws + 8  * MB);    // 32 MB bf16 U
    float*          hf  = (float*)(ws + 40 * MB);             //  8 MB f32 h
    unsigned short* hb  = (unsigned short*)(ws + 48 * MB);    //  4 MB bf16 h
    float*          Mm  = (float*)(ws + 52 * MB);             //  4 MB folded Wq*keys
    float*          simb= (float*)(ws + 56 * MB);             //  8 MB sim
    int*            eix = (int*)(ws + 64 * MB);               // 256 KB indices
    float*          ewt = (float*)(ws + 64 * MB + 256u * 1024u); // 256 KB weights

    // K0: X -> bf16   (4M elems / 4 per thread)
    f32_to_bf16_kernel<<<4096, 256, 0, stream>>>(X, (unsigned int*)Xb, 2048 * 2048 / 4);

    // K_M: fold W_q @ keys  -> Mm [1024,1024]
    build_m_kernel<<<dim3(1024, 4), 256, 0, stream>>>(Wq, keys, Mm);

    // GEMM1: U = silu(X @ Wup)  [2048,8192], bf16 out
    gemm_bf16_wmma<2048, 8192, 2048, true, false, true>
        <<<dim3(32, 16), 256, 0, stream>>>(Xb, Wup, nullptr, U);

    // GEMM2: h = U @ Wdn  [2048,1024], f32 + bf16 out
    gemm_bf16_wmma<2048, 1024, 8192, false, true, true>
        <<<dim3(4, 16), 256, 0, stream>>>(U, Wdn, hf, hb);

    // GEMM3: sim = h @ Mm  [2048,1024], f32 out
    gemm_bf16_wmma<2048, 1024, 1024, false, true, false>
        <<<dim3(4, 16), 256, 0, stream>>>(hb, Mm, simb, nullptr);

    // K4: two-stage top-k + softmax  (8192 threads)
    topk_kernel<<<32, 256, 0, stream>>>(simb, eix, ewt);

    // K5: expert readout
    moe_out_kernel<<<2048, 256, 0, stream>>>(hf, upE, dnE, eix, ewt, out);
}
